// PointTransformerLayer_27127013441595
// MI455X (gfx1250) — compile-verified, matrix operand order checked by reading, not run on hardware
//
#include <hip/hip_runtime.h>
#include <hip/hip_fp16.h>

// ---------------------------------------------------------------------------
// Point Transformer layer for MI455X (gfx1250, wave32, WMMA + TDM).
// B=2, N=M=8192, K=16, CIN=H=COUT=128, G=8.
//
//  k0  : convert 7 weight matrices to f16, zero groupnorm stat cells
//  k1  : query/key/value conv1d  (WMMA GEMM 128x128 over 8192 pts)
//  k2  : pe1 = d1_w * rel + b (K=3 dot, VALU) + d-GN stats (atomics)
//  k3  : fold stats -> per-channel scale/shift
//  k4  : pe = d2_w * relu(gn(pe1)) + b             (TDM stage + WMMA)
//  k5  : a1 = g1_w * (q - gather(k) + pe) + b      (TDM stage + WMMA) + stats
//  k3  : fold g stats
//  k7  : a2 GEMM + mask + softmax(K) + sum_k a2*(v+pe)  (2x TDM + WMMA)
//  k8  : out = post_w * result + post_b + identity (2D TDM + WMMA, NT store)
// ---------------------------------------------------------------------------

typedef __attribute__((ext_vector_type(16))) _Float16 v16h;
typedef __attribute__((ext_vector_type(8)))  _Float16 v8h;
typedef __attribute__((ext_vector_type(8)))  float    v8f;
typedef __attribute__((ext_vector_type(4)))  unsigned int u32x4;
typedef __attribute__((ext_vector_type(8)))  int          i32x8;
typedef __attribute__((ext_vector_type(4)))  int          i32x4;

#define Bn 2
#define Np 8192
#define Mp 8192
#define Kn 16
#define Cn 128
#define Gn 8

#define HAVE_TDM __has_builtin(__builtin_amdgcn_tensor_load_to_lds)

// ---- Tensor Data Mover ----------------------------------------------------
#if HAVE_TDM
__device__ __forceinline__ unsigned lds_off(const void* p) {
  // generic LDS address: low 32 bits are the LDS byte offset (aperture calc)
  return (unsigned)(unsigned long long)(uintptr_t)p;
}

// 2D tile load: tile0 x tile1 elements of (1<<ds_code) bytes, row stride
// stride0 elements. Descriptor per CDNA5 ISA.
// This toolchain exposes the 6-arg builtin:
//   (u32x4 g0, i32x8 g1, i32x4 g2, i32x4 g3, i32x8 g4, i32 cpol)
__device__ __forceinline__ void tdm_issue(const void* gsrc, unsigned ldsoff,
                                          unsigned ds_code, unsigned tile0,
                                          unsigned tile1, unsigned tdim0,
                                          unsigned tdim1,
                                          unsigned long long stride0) {
  unsigned long long ga = (unsigned long long)(uintptr_t)gsrc;
  u32x4 g0 = {1u,                                   // count=1 (user D#)
              ldsoff,                                // lds_addr
              (unsigned)ga,                          // global_addr[95:64]
              ((unsigned)((ga >> 32) & 0x01ffffffu)) | 0x80000000u}; // type=2
  i32x8 g1;
  g1[0] = (int)(ds_code << 16);                      // mask=0, data_size
  g1[1] = (int)((tdim0 & 0xffffu) << 16);            // tensor_dim0[15:0]
  g1[2] = (int)(((tdim0 >> 16) & 0xffffu) | ((tdim1 & 0xffffu) << 16));
  g1[3] = (int)(((tdim1 >> 16) & 0xffffu) | (tile0 << 16));  // tile_dim0
  g1[4] = (int)(tile1 & 0xffffu);                    // tile_dim1, tile_dim2=0
  g1[5] = (int)(unsigned)(stride0 & 0xffffffffu);    // tensor_dim0_stride lo
  g1[6] = (int)(unsigned)((stride0 >> 32) & 0xffffu);
  g1[7] = 0;
  i32x4 gz4 = {0, 0, 0, 0};
  i32x8 gz8 = {0, 0, 0, 0, 0, 0, 0, 0};
  __builtin_amdgcn_tensor_load_to_lds(g0, g1, gz4, gz4, gz8, 0);
}
#endif

// ---- WMMA helpers ---------------------------------------------------------
__device__ __forceinline__ v16h load_a16(const _Float16* __restrict__ W,
                                         int row, int c0, int h) {
  const _Float16* p = W + row * 128 + c0 + (h << 3);
  v8h lo = *(const v8h*)p;
  v8h hi = *(const v8h*)(p + 16);
  return __builtin_shufflevector(lo, hi, 0, 1, 2, 3, 4, 5, 6, 7,
                                 8, 9, 10, 11, 12, 13, 14, 15);
}

__device__ __forceinline__ int bop_idx(int c, int n) {
  int t = c >> 5, c5 = c & 31, h, j;
  if (c5 < 16) { h = c5 >> 3; j = c5 & 7; }
  else         { h = (c5 - 16) >> 3; j = 8 + ((c5 - 16) & 7); }
  return ((t << 5) + n + (h << 4)) * 16 + j;
}

template <int NS>
__device__ __forceinline__ void gemm_tiles(const _Float16* __restrict__ W,
                                           const _Float16* __restrict__ sB,
                                           int lane, int obase, v8f acc[NS]) {
  int row = obase + (lane & 15);
  int h = lane >> 4;
#pragma unroll
  for (int t = 0; t < 4; ++t) {
    v16h a = load_a16(W, row, t * 32, h);
#pragma unroll
    for (int s = 0; s < NS; ++s) {
      v16h b = *(const v16h*)(sB + s * 2048 + ((t << 5) + lane) * 16);
      acc[s] = __builtin_amdgcn_wmma_f32_16x16x32_f16(
          false, a, false, b, (short)0, acc[s], false, false);
    }
  }
}

// ---- wave32 shuffle reductions --------------------------------------------
__device__ __forceinline__ float red16_add(float v) {
#pragma unroll
  for (int m = 1; m < 16; m <<= 1) v += __shfl_xor(v, m, 32);
  return v;
}
__device__ __forceinline__ float red16_max(float v) {
#pragma unroll
  for (int m = 1; m < 16; m <<= 1) v = fmaxf(v, __shfl_xor(v, m, 32));
  return v;
}
__device__ __forceinline__ float red32_add(float v) {
#pragma unroll
  for (int m = 1; m < 32; m <<= 1) v += __shfl_xor(v, m, 32);
  return v;
}

// ---- k0: weight conversion + stat zeroing ---------------------------------
__global__ __launch_bounds__(256) void k0_prep(
    const float* __restrict__ wq, const float* __restrict__ wk,
    const float* __restrict__ wv, const float* __restrict__ d2,
    const float* __restrict__ g1, const float* __restrict__ g2,
    const float* __restrict__ pw, _Float16* __restrict__ wf16,
    float* __restrict__ stats) {
  const float* src[7] = {wq, wk, wv, d2, g1, g2, pw};
  int m = blockIdx.x;
  const float* s = src[m];
  _Float16* d = wf16 + m * 16384;
  for (int i = threadIdx.x; i < 16384; i += 256) d[i] = (_Float16)s[i];
  if (m == 0 && threadIdx.x < 64) stats[threadIdx.x] = 0.f;
}

// ---- k1: q/k/v conv1d -----------------------------------------------------
__global__ __launch_bounds__(256) void k1_qkv(
    const float* __restrict__ qf, const float* __restrict__ kf,
    const float* __restrict__ vf, const _Float16* __restrict__ wf16,
    const float* __restrict__ bq, const float* __restrict__ bk,
    const float* __restrict__ bv, float* __restrict__ query,
    float* __restrict__ keyB, float* __restrict__ valB) {
  __shared__ __align__(32) _Float16 sB[4 * 2048];
  int bx = blockIdx.x;             // 3 * Bn * 128 = 768
  int mat = bx >> 8;
  int rem = bx & 255;
  int b = rem >> 7;
  int n0 = (rem & 127) << 6;
  const float* X = mat == 0 ? qf : (mat == 1 ? kf : vf);
  const float* bias = mat == 0 ? bq : (mat == 1 ? bk : bv);
  float* O = mat == 0 ? query : (mat == 1 ? keyB : valB);
  const _Float16* W = wf16 + mat * 16384;
  int tid = threadIdx.x;
#pragma unroll
  for (int i = 0; i < 32; ++i) {
    int e = i * 256 + tid;
    int c = e >> 6, n = e & 63;
    float v = X[(b * Cn + c) * Np + n0 + n];
    sB[(n >> 4) * 2048 + bop_idx(c, n & 15)] = (_Float16)v;
  }
  __syncthreads();
  int lane = tid & 31, w = tid >> 5;
  v8f acc[4] = {};
  gemm_tiles<4>(W, sB, lane, w * 16, acc);
  int col = lane & 15, hh = lane >> 4;
#pragma unroll
  for (int s = 0; s < 4; ++s)
#pragma unroll
    for (int r = 0; r < 8; ++r) {
      int row = w * 16 + r + (hh << 3);
      O[(b * Cn + row) * Np + n0 + s * 16 + col] = acc[s][r] + bias[row];
    }
}

// ---- k2: pe1 (K=3 conv) + d-GN stats --------------------------------------
__global__ __launch_bounds__(256) void k2_pe1(
    const float* __restrict__ qx, const float* __restrict__ kx,
    const int* __restrict__ knn, const float* __restrict__ d1w,
    const float* __restrict__ d1b, _Float16* __restrict__ pe1,
    float* __restrict__ statsD) {
  __shared__ float rel[4][3][16];
  __shared__ float sgrp[16];
  int bx = blockIdx.x;             // 4096
  int b = bx >> 11;
  int m0 = (bx & 2047) << 2;
  int tid = threadIdx.x;
  if (tid < 16) sgrp[tid] = 0.f;
  if (tid < 192) {
    int ml = tid / 48, rr = tid % 48, d = rr >> 4, k = rr & 15;
    int m = m0 + ml;
    int idx = knn[(b * Mp + m) * Kn + k];
    rel[ml][d][k] = qx[(b * 3 + d) * Mp + m] - kx[(b * 3 + d) * Np + idx];
  }
  __syncthreads();
  int c = tid & 127, half = tid >> 7;
  float w0 = d1w[c * 3 + 0], w1 = d1w[c * 3 + 1], w2 = d1w[c * 3 + 2];
  float bb = d1b[c];
  float s1 = 0.f, s2 = 0.f;
  for (int ml = 0; ml < 4; ++ml) {
    int m = m0 + ml;
    _Float16* dst = pe1 + ((size_t)(b * Mp + m) * Cn + c) * Kn;
#pragma unroll
    for (int kk = 0; kk < 8; ++kk) {
      int k = half * 8 + kk;
      float v = w0 * rel[ml][0][k] + w1 * rel[ml][1][k] + w2 * rel[ml][2][k] + bb;
      dst[k] = (_Float16)v;
      s1 += v; s2 += v * v;
    }
  }
  int g = c >> 4;
  atomicAdd(&sgrp[g * 2 + 0], s1);
  atomicAdd(&sgrp[g * 2 + 1], s2);
  __syncthreads();
  if (tid < 16) atomicAdd(&statsD[b * 16 + tid], sgrp[tid]);
}

// ---- k3: fold GN stats into per-channel scale/shift -----------------------
__global__ __launch_bounds__(256) void k3_gnfin(
    const float* __restrict__ stats, const float* __restrict__ gamma,
    const float* __restrict__ beta, float* __restrict__ ss, float invcnt) {
  int t = threadIdx.x;             // Bn*Cn = 256
  int b = t >> 7, c = t & 127, g = c >> 4;
  float s1 = stats[(b * Gn + g) * 2 + 0];
  float s2 = stats[(b * Gn + g) * 2 + 1];
  float mu = s1 * invcnt;
  float var = s2 * invcnt - mu * mu;
  float rs = rsqrtf(var + 1e-5f);
  float sc = gamma[c] * rs;
  ss[(b * Cn + c) * 2 + 0] = sc;
  ss[(b * Cn + c) * 2 + 1] = beta[c] - mu * sc;
}

// ---- staging helper: contiguous 16KB f16 tile -> LDS ----------------------
__device__ __forceinline__ void stage_tile(const _Float16* __restrict__ src,
                                           _Float16* __restrict__ stg, int tid) {
#if HAVE_TDM
  if (tid < 32) {   // one wave drives the DMA
    tdm_issue(src, lds_off(stg), /*8B elems*/3, 2048, 1, 2048, 1, 2048ull);
    __builtin_amdgcn_s_wait_tensorcnt(0);
  }
#else
  for (int i = 0; i < 32; ++i) {
    int e = i * 256 + tid;
    stg[e] = src[e];
  }
#endif
}

// ---- k4: pe = d2_w * relu(gn(pe1)) + b ------------------------------------
__global__ __launch_bounds__(256) void k4_pegemm(
    const _Float16* __restrict__ pe1, const float* __restrict__ dss,
    const _Float16* __restrict__ wf16, const float* __restrict__ d2b,
    _Float16* __restrict__ peF) {
  __shared__ __align__(32) _Float16 sB[4 * 2048];
  __shared__ __align__(128) _Float16 stg[8192];
  int bx = blockIdx.x;             // 4096
  int b = bx >> 11;
  int m0 = (bx & 2047) << 2;
  int tid = threadIdx.x;
  stage_tile(pe1 + (size_t)(b * Mp + m0) * (Cn * Kn), stg, tid);
  __syncthreads();
#pragma unroll
  for (int i = 0; i < 32; ++i) {
    int e = i * 256 + tid;
    int ml = e >> 11, c = (e >> 4) & 127, k = e & 15;
    float x = (float)stg[e];
    float sc = dss[(b * Cn + c) * 2], sh = dss[(b * Cn + c) * 2 + 1];
    x = fmaxf(sc * x + sh, 0.f);
    sB[ml * 2048 + bop_idx(c, k)] = (_Float16)x;
  }
  __syncthreads();
  int lane = tid & 31, w = tid >> 5;
  v8f acc[4] = {};
  gemm_tiles<4>(wf16 + 3 * 16384, sB, lane, w * 16, acc);
  int col = lane & 15, hh = lane >> 4;
#pragma unroll
  for (int s = 0; s < 4; ++s) {
    int m = m0 + s;
#pragma unroll
    for (int r = 0; r < 8; ++r) {
      int row = w * 16 + r + (hh << 3);
      peF[((size_t)(b * Mp + m) * Cn + row) * Kn + col] =
          (_Float16)(acc[s][r] + d2b[row]);
    }
  }
}

// ---- k5: a1 = g1_w * (query - gather(key) + pe) + b ; g-GN stats ----------
__global__ __launch_bounds__(256) void k5_attn_in(
    const float* __restrict__ query, const float* __restrict__ keyB,
    const _Float16* __restrict__ peF, const int* __restrict__ knn,
    const _Float16* __restrict__ wf16, const float* __restrict__ g1b,
    _Float16* __restrict__ a1, float* __restrict__ statsG) {
  __shared__ __align__(32) _Float16 sB[4 * 2048];
  __shared__ __align__(128) _Float16 stg[8192];
  __shared__ int sidx[4][16];
  int bx = blockIdx.x;             // 4096
  int b = bx >> 11;
  int m0 = (bx & 2047) << 2;
  int tid = threadIdx.x;
  // TDM streams the pe tile into LDS while lanes fetch KNN indices
  stage_tile(peF + (size_t)(b * Mp + m0) * (Cn * Kn), stg, tid);
  if (tid < 64) {
    int ml = tid >> 4, k = tid & 15;
    int idx = knn[(b * Mp + m0 + ml) * Kn + k];
    sidx[ml][k] = idx;
    __builtin_prefetch(&keyB[(b * Cn) * Np + idx], 0, 1);  // global_prefetch_b8
  }
  __syncthreads();
#pragma unroll
  for (int i = 0; i < 32; ++i) {
    int e = i * 256 + tid;
    int ml = e >> 11, c = (e >> 4) & 127, k = e & 15;
    int m = m0 + ml;
    float v = query[(b * Cn + c) * Mp + m]
            - keyB[(b * Cn + c) * Np + sidx[ml][k]]
            + (float)stg[e];
    sB[ml * 2048 + bop_idx(c, k)] = (_Float16)v;
  }
  __syncthreads();
  int lane = tid & 31, w = tid >> 5;
  v8f acc[4] = {};
  gemm_tiles<4>(wf16 + 4 * 16384, sB, lane, w * 16, acc);
  int col = lane & 15, hh = lane >> 4;
  float s1 = 0.f, s2 = 0.f;
#pragma unroll
  for (int s = 0; s < 4; ++s) {
    int m = m0 + s;
#pragma unroll
    for (int r = 0; r < 8; ++r) {
      int row = w * 16 + r + (hh << 3);
      float x = acc[s][r] + g1b[row];
      a1[((size_t)(b * Mp + m) * Cn + row) * Kn + col] = (_Float16)x;
      s1 += x; s2 += x * x;
    }
  }
  s1 = red32_add(s1);
  s2 = red32_add(s2);
  if (lane == 0) {      // wave w's 16 rows == GN group w
    atomicAdd(&statsG[(b * Gn + w) * 2 + 0], s1);
    atomicAdd(&statsG[(b * Gn + w) * 2 + 1], s2);
  }
}

// ---- k7: a2 GEMM + mask + softmax(K) + sum_k a2*(v+pe) --------------------
__global__ __launch_bounds__(256) void k7_attn(
    const _Float16* __restrict__ a1, const float* __restrict__ gss,
    const _Float16* __restrict__ wf16, const float* __restrict__ g2b,
    const unsigned char* __restrict__ mask, const int* __restrict__ knn,
    const float* __restrict__ valB, const _Float16* __restrict__ peF,
    _Float16* __restrict__ resf) {
  __shared__ __align__(32) _Float16 sB[4 * 2048];
  __shared__ __align__(128) _Float16 stgA[8192];
  __shared__ __align__(128) _Float16 stgP[8192];
  int bx = blockIdx.x;             // 4096
  int b = bx >> 11;
  int m0 = (bx & 2047) << 2;
  int tid = threadIdx.x;
#if HAVE_TDM
  if (tid < 32) {                  // two descriptors in flight, one wait
    tdm_issue(a1 + (size_t)(b * Mp + m0) * (Cn * Kn), lds_off(stgA),
              3, 2048, 1, 2048, 1, 2048ull);
    tdm_issue(peF + (size_t)(b * Mp + m0) * (Cn * Kn), lds_off(stgP),
              3, 2048, 1, 2048, 1, 2048ull);
    __builtin_amdgcn_s_wait_tensorcnt(0);
  }
#else
  for (int i = 0; i < 32; ++i) {
    int e = i * 256 + tid;
    stgA[e] = a1[(size_t)(b * Mp + m0) * (Cn * Kn) + e];
    stgP[e] = peF[(size_t)(b * Mp + m0) * (Cn * Kn) + e];
  }
#endif
  __syncthreads();
#pragma unroll
  for (int i = 0; i < 32; ++i) {
    int e = i * 256 + tid;
    int ml = e >> 11, c = (e >> 4) & 127, k = e & 15;
    float x = (float)stgA[e];
    float sc = gss[(b * Cn + c) * 2], sh = gss[(b * Cn + c) * 2 + 1];
    x = fmaxf(sc * x + sh, 0.f);
    sB[ml * 2048 + bop_idx(c, k)] = (_Float16)x;
  }
  __syncthreads();
  int lane = tid & 31, w = tid >> 5;
  v8f acc[4] = {};
  gemm_tiles<4>(wf16 + 5 * 16384, sB, lane, w * 16, acc);
  const float invs = 0.08838834764831845f;   // 1/sqrt(128)
  int col = lane & 15, hh = lane >> 4;
#pragma unroll
  for (int s = 0; s < 4; ++s) {
    int m = m0 + s;
    int idx = knn[(b * Mp + m) * Kn + col];
    bool mk = mask[(b * Mp + m) * Kn + col] != 0;
    float out[8];
#pragma unroll
    for (int r = 0; r < 8; ++r) {
      int row = w * 16 + r + (hh << 3);
      float x = (acc[s][r] + g2b[row]) * invs;
      x = mk ? x : -1e30f;
      float mx = red16_max(x);          // softmax over K=16 (cols = lanes)
      float e = __expf(x - mx);
      float sum = red16_add(e);
      float wgt = e / sum;
      float vp = valB[(b * Cn + row) * Np + idx]
               + (float)stgP[(s << 11) + row * 16 + col];
      out[r] = red16_add(wgt * vp);
    }
    if (col == 0) {
#pragma unroll
      for (int r = 0; r < 8; ++r) {
        int row = w * 16 + r + (hh << 3);
        resf[(b * Cn + row) * Mp + m] = (_Float16)out[r];
      }
    }
  }
}

// ---- k8: out = post_w * result + post_b + identity ------------------------
__global__ __launch_bounds__(256) void k8_post(
    const _Float16* __restrict__ resf, const _Float16* __restrict__ wf16,
    const float* __restrict__ pb, const float* __restrict__ qf,
    float* __restrict__ out) {
  __shared__ __align__(32) _Float16 sB[4 * 2048];
  __shared__ __align__(128) _Float16 stg[8192];
  int bx = blockIdx.x;             // Bn*128 = 256
  int b = bx >> 7;
  int n0 = (bx & 127) << 6;
  int tid = threadIdx.x;
  const _Float16* src = resf + (size_t)(b * Cn) * Mp + n0;
#if HAVE_TDM
  if (tid < 32) {                  // strided 2D tile: 128 rows x 64 f16
    tdm_issue(src, lds_off(stg), /*2B elems*/1, 64, 128,
              1u << 20, 1u << 20, (unsigned long long)Mp);
    __builtin_amdgcn_s_wait_tensorcnt(0);
  }
#else
  for (int i = 0; i < 32; ++i) {
    int e = i * 256 + tid;
    int c = e >> 6, n = e & 63;
    stg[e] = src[(size_t)c * Mp + n];
  }
#endif
  __syncthreads();
#pragma unroll
  for (int i = 0; i < 32; ++i) {
    int e = i * 256 + tid;
    int c = e >> 6, n = e & 63;
    sB[(n >> 4) * 2048 + bop_idx(c, n & 15)] = stg[e];
  }
  __syncthreads();
  int lane = tid & 31, w = tid >> 5;
  v8f acc[4] = {};
  gemm_tiles<4>(wf16 + 6 * 16384, sB, lane, w * 16, acc);
  int col = lane & 15, hh = lane >> 4;
#pragma unroll
  for (int s = 0; s < 4; ++s)
#pragma unroll
    for (int r = 0; r < 8; ++r) {
      int row = w * 16 + r + (hh << 3);
      int o = (b * Cn + row) * Mp + n0 + s * 16 + col;
      float v = acc[s][r] + pb[row] + qf[o];
      __builtin_nontemporal_store(v, &out[o]);   // write-once, keep L2 clean
    }
}

// ---------------------------------------------------------------------------
extern "C" void kernel_launch(void* const* d_in, const int* in_sizes, int n_in,
                              void* d_out, int out_size, void* d_ws,
                              size_t ws_size, hipStream_t stream) {
  const float* q_xyzs  = (const float*)d_in[0];
  const float* k_xyzs  = (const float*)d_in[1];
  const float* q_feats = (const float*)d_in[2];
  const float* k_feats = (const float*)d_in[3];
  const float* v_feats = (const float*)d_in[4];
  const int*   knn     = (const int*)d_in[5];
  const unsigned char* mask = (const unsigned char*)d_in[6];
  const float* wq_w = (const float*)d_in[7];
  const float* wq_b = (const float*)d_in[8];
  const float* wk_w = (const float*)d_in[9];
  const float* wk_b = (const float*)d_in[10];
  const float* wv_w = (const float*)d_in[11];
  const float* wv_b = (const float*)d_in[12];
  const float* d1_w = (const float*)d_in[13];
  const float* d1_b = (const float*)d_in[14];
  const float* d_gn_g = (const float*)d_in[15];
  const float* d_gn_b = (const float*)d_in[16];
  const float* d2_w = (const float*)d_in[17];
  const float* d2_b = (const float*)d_in[18];
  const float* g1_w = (const float*)d_in[19];
  const float* g1_b = (const float*)d_in[20];
  const float* g_gn_g = (const float*)d_in[21];
  const float* g_gn_b = (const float*)d_in[22];
  const float* g2_w = (const float*)d_in[23];
  const float* g2_b = (const float*)d_in[24];
  const float* post_w = (const float*)d_in[25];
  const float* post_b = (const float*)d_in[26];
  float* out = (float*)d_out;

  char* base = (char*)d_ws;
  size_t off = 0;
  auto nxt = [&](size_t bytes) {
    char* p = base + off;
    off = (off + bytes + 255) & ~(size_t)255;
    return p;
  };
  _Float16* wf16  = (_Float16*)nxt((size_t)7 * 16384 * 2);
  float*    query = (float*)nxt((size_t)Bn * Cn * Mp * 4);
  float*    keyB  = (float*)nxt((size_t)Bn * Cn * Np * 4);
  float*    valB  = (float*)nxt((size_t)Bn * Cn * Np * 4);
  _Float16* pe1a1 = (_Float16*)nxt((size_t)Bn * Mp * Cn * Kn * 2);  // pe1 -> a1
  _Float16* peF   = (_Float16*)nxt((size_t)Bn * Mp * Cn * Kn * 2);
  _Float16* resf  = (_Float16*)nxt((size_t)Bn * Cn * Mp * 2);
  float*    dss   = (float*)nxt((size_t)Bn * Cn * 2 * 4);
  float*    gss   = (float*)nxt((size_t)Bn * Cn * 2 * 4);
  float*    stats = (float*)nxt(64 * 4);
  float* statsD = stats;
  float* statsG = stats + 32;
  (void)in_sizes; (void)n_in; (void)out_size; (void)ws_size;

  const float invcnt = 1.0f / (float)((Cn / Gn) * Mp * Kn);  // 1/2097152

  k0_prep<<<7, 256, 0, stream>>>(wq_w, wk_w, wv_w, d2_w, g1_w, g2_w, post_w,
                                 wf16, stats);
  k1_qkv<<<768, 256, 0, stream>>>(q_feats, k_feats, v_feats, wf16,
                                  wq_b, wk_b, wv_b, query, keyB, valB);
  k2_pe1<<<4096, 256, 0, stream>>>(q_xyzs, k_xyzs, knn, d1_w, d1_b,
                                   pe1a1, statsD);
  k3_gnfin<<<1, 256, 0, stream>>>(statsD, d_gn_g, d_gn_b, dss, invcnt);
  k4_pegemm<<<4096, 256, 0, stream>>>(pe1a1, dss, wf16, d2_b, peF);
  k5_attn_in<<<4096, 256, 0, stream>>>(query, keyB, peF, knn, wf16, g1_b,
                                       pe1a1 /* now a1 */, statsG);
  k3_gnfin<<<1, 256, 0, stream>>>(statsG, g_gn_g, g_gn_b, gss, invcnt);
  k7_attn<<<4096, 256, 0, stream>>>(pe1a1 /* a1 */, gss, wf16, g2_b, mask,
                                    knn, valB, peF, resf);
  k8_post<<<256, 256, 0, stream>>>(resf, wf16, post_b, q_feats, out);
}
